// StatePoolLayer_91122026152399
// MI455X (gfx1250) — matrix operand order repaired and checked by reading, not compiled, not run on hardware
//
#include <hip/hip_runtime.h>
#include <hip/hip_bf16.h>
#include <math.h>

// ---------------------------------------------------------------------------
// Problem constants (match the reference)
// ---------------------------------------------------------------------------
#define RR     8192
#define A14    14
#define ATOMS  (RR * A14)          // 114688
#define XX     3
#define NN0    256
#define NN1    256
#define NHH    8

typedef __attribute__((ext_vector_type(16))) _Float16 v16h;
typedef __attribute__((ext_vector_type(8)))  _Float16 v8h;
typedef __attribute__((ext_vector_type(4)))  _Float16 v4h;
typedef __attribute__((ext_vector_type(8)))  float    v8f;
typedef __attribute__((ext_vector_type(4)))  float    v4f;
typedef __attribute__((ext_vector_type(4)))  int      v4i;

// ---- CDNA5 async global->LDS copy path (guarded; fallback always compiles) --
#ifndef USE_ASYNC_LDS
#if defined(__has_builtin)
#if __has_builtin(__builtin_amdgcn_global_load_async_to_lds_b128) && \
    __has_builtin(__builtin_amdgcn_s_wait_asynccnt)
#define USE_ASYNC_LDS 1
#endif
#endif
#endif
#ifndef USE_ASYNC_LDS
#define USE_ASYNC_LDS 0
#endif

#if USE_ASYNC_LDS
// builtin signature (probe-verified): param0 = int4 addrspace(1)*,
// param1 = int4 addrspace(3)*, then two immediate ints (offset, cpol)
#define AS1P(p) ((__attribute__((address_space(1))) v4i*)(p))
#define AS3P(p) ((__attribute__((address_space(3))) v4i*)(p))
#endif

// ---------------------------------------------------------------------------
// Weight repack: W[K,N] f32 row-major  ->  WMMA B-fragment layout, f16.
// For each 32-K chunk kc and 16-col tile nt:
//   lane 0..15  : col = nt*16+lane,    K = kc*32 + 0..15   (16 contiguous f16)
//   lane 16..31 : col = nt*16+lane-16, K = kc*32 + 16..31
// Packed offset: ((kc*(N/16)+nt)*32 + lane)*16 halves.
// A 32-K chunk of NT consecutive tiles is one contiguous NT*1KB region.
// ---------------------------------------------------------------------------
__global__ void pack_b_kernel(const float* __restrict__ W,
                              _Float16* __restrict__ Bp,
                              int K, int N) {
    const int lane   = threadIdx.x;           // 32 threads
    const int tile   = blockIdx.x;            // kc*(N/16)+nt
    const int ntiles = N >> 4;
    const int kc     = tile / ntiles;
    const int nt     = tile - kc * ntiles;
    const int col    = nt * 16 + (lane & 15);
    const int kb     = kc * 32 + ((lane >> 4) * 16);

    _Float16 tmp[16];
#pragma unroll
    for (int j = 0; j < 16; ++j)
        tmp[j] = (_Float16)W[(size_t)(kb + j) * N + col];

    _Float16* dst = Bp + ((size_t)tile * 32 + lane) * 16;
    *(v8h*)(dst)     = *(v8h*)(tmp);
    *(v8h*)(dst + 8) = *(v8h*)(tmp + 8);
}

// ---------------------------------------------------------------------------
// Prep: z = concat(q, sqrt(sum_x p^2 + 1e-6)) in f16, plus f16 copies of q, p.
// One thread = 4 consecutive channels of one atom (b128 loads, b64 stores).
// ---------------------------------------------------------------------------
__global__ __launch_bounds__(256)
void prep_kernel(const float* __restrict__ q,
                 const float* __restrict__ p,
                 _Float16* __restrict__ z16,
                 _Float16* __restrict__ q16,
                 _Float16* __restrict__ p16) {
    const size_t t  = (size_t)blockIdx.x * 256u + threadIdx.x;  // ATOMS*64 total
    const size_t a  = t >> 6;
    const int    c4 = ((int)(t & 63)) << 2;

    v4f qv = *(const v4f*)(q + a * NN0 + c4);
    v4f p0 = *(const v4f*)(p + (a * 3 + 0) * NN0 + c4);
    v4f p1 = *(const v4f*)(p + (a * 3 + 1) * NN0 + c4);
    v4f p2 = *(const v4f*)(p + (a * 3 + 2) * NN0 + c4);

    v4h qh, nh, h0, h1, h2;
#pragma unroll
    for (int i = 0; i < 4; ++i) {
        float ss = p0[i] * p0[i] + p1[i] * p1[i] + p2[i] * p2[i];
        float nv = sqrtf(ss + 1e-6f);                 // _sqrt: arg >= 0 here
        qh[i] = (_Float16)qv[i];
        nh[i] = (_Float16)nv;
        h0[i] = (_Float16)p0[i];
        h1[i] = (_Float16)p1[i];
        h2[i] = (_Float16)p2[i];
    }
    *(v4h*)(z16 + a * 512 + c4)           = qh;
    *(v4h*)(z16 + a * 512 + 256 + c4)     = nh;
    *(v4h*)(q16 + a * NN0 + c4)           = qh;
    *(v4h*)(p16 + (a * 3 + 0) * NN0 + c4) = h0;
    *(v4h*)(p16 + (a * 3 + 1) * NN0 + c4) = h1;
    *(v4h*)(p16 + (a * 3 + 2) * NN0 + c4) = h2;
}

// ---------------------------------------------------------------------------
// Generic WMMA GEMM: C[M,N] = act(A[M,K] * B[K,N] + bias)
//   - A: f16 row-major;  B: pre-packed f16 fragments (pack_b_kernel)
//   - 256 threads = 8 waves; block tile (MT*128) x (NT*16)
//   - wave tile (MT*16) rows x (NT*16) cols -> MT*NT WMMAs per 32-K chunk
//   - B chunks staged in LDS (double-buffered, async global->LDS)
//   - A fragments double-buffered in registers (issue k+1 before k's WMMAs)
//   - all B frags loaded from LDS before WMMA block -> uninterrupted MMA burst
//   - f32 accumulation via v_wmma_f32_16x16x32_f16
// ---------------------------------------------------------------------------
template <int MT, int NT, bool BIAS, bool RELU, bool OUT_F16>
__global__ __launch_bounds__(256)
void wmma_gemm_kernel(const _Float16* __restrict__ A,
                      const _Float16* __restrict__ Bp,
                      const float*    __restrict__ bias,
                      void*           __restrict__ Cout,
                      int M, int K, int N) {
    __shared__ __align__(16) _Float16 sB[2][NT * 512];

    const int tid     = threadIdx.x;
    const int lane    = tid & 31;
    const int wave    = tid >> 5;
    const int rowbase = blockIdx.x * (MT * 128) + wave * (MT * 16);
    const int colg    = blockIdx.y;                 // group of NT*16 columns
    const int colbase = colg * (NT * 16);
    const int r       = lane & 15;
    const int hiHalf  = lane >> 4;                  // 0 | 1
    const int ntilesN = N >> 4;
    const int kchunks = K >> 5;
    const int copyThreads = NT * 64;                // chunk = NT*512 halves, 16B/thr

    // ---- B chunk copy: global (packed, contiguous per chunk) -> LDS --------
    auto copy_chunk = [&](int kc, int buf) {
        if (tid < copyThreads) {
            const _Float16* src =
                Bp + ((size_t)(kc * ntilesN + colg * NT) * 512) + tid * 8;
            _Float16* dst = &sB[buf][tid * 8];
#if USE_ASYNC_LDS
            __builtin_amdgcn_global_load_async_to_lds_b128(AS1P(src), AS3P(dst), 0, 0);
#else
            *(v8h*)dst = *(const v8h*)src;
#endif
        }
    };

    v8f acc[MT][NT] = {};
    const _Float16* Arow[MT];
#pragma unroll
    for (int m = 0; m < MT; ++m)
        Arow[m] = A + (size_t)(rowbase + m * 16 + r) * K + hiHalf * 16;

    // ---- prologue: chunk 0 B into LDS, chunk 0 A into registers ------------
    copy_chunk(0, 0);
    v8h alo[MT], ahi[MT];
#pragma unroll
    for (int m = 0; m < MT; ++m) {
        alo[m] = *(const v8h*)(Arow[m]);
        ahi[m] = *(const v8h*)(Arow[m] + 8);
    }
#if USE_ASYNC_LDS
    __builtin_amdgcn_s_wait_asynccnt(0);
#endif
    __syncthreads();

    for (int kc = 0; kc < kchunks; ++kc) {
        const int cur = kc & 1;
        const bool more = (kc + 1 < kchunks);

        if (more) copy_chunk(kc + 1, cur ^ 1);      // overlaps this chunk's math

        v8h alo_n[MT], ahi_n[MT];
#pragma unroll
        for (int m = 0; m < MT; ++m) { alo_n[m] = alo[m]; ahi_n[m] = ahi[m]; }
        if (more) {
#pragma unroll
            for (int m = 0; m < MT; ++m) {
                const _Float16* ap = Arow[m] + (kc + 1) * 32;
                alo_n[m] = *(const v8h*)(ap);
                ahi_n[m] = *(const v8h*)(ap + 8);
            }
            if (kc + 3 < kchunks)
                __builtin_prefetch(Arow[0] + (kc + 3) * 32, 0, 0);  // global_prefetch_b8
        }

        // ---- all B fragments for this chunk first (one dscnt wait) ---------
        v16h bfrag[NT];
#pragma unroll
        for (int t = 0; t < NT; ++t) {
            const _Float16* bp = &sB[cur][t * 512 + lane * 16];  // ds_load_b128 x2
            v8h blo = *(const v8h*)(bp);
            v8h bhi = *(const v8h*)(bp + 8);
#pragma unroll
            for (int i = 0; i < 8; ++i) { bfrag[t][i] = blo[i]; bfrag[t][8 + i] = bhi[i]; }
        }

        v16h afrag[MT];
#pragma unroll
        for (int m = 0; m < MT; ++m)
#pragma unroll
            for (int i = 0; i < 8; ++i) { afrag[m][i] = alo[m][i]; afrag[m][8 + i] = ahi[m][i]; }

        // ---- MT*NT back-to-back WMMAs (independent accumulators) -----------
#pragma unroll
        for (int m = 0; m < MT; ++m)
#pragma unroll
            for (int t = 0; t < NT; ++t)
                acc[m][t] = __builtin_amdgcn_wmma_f32_16x16x32_f16(
                    false, afrag[m], false, bfrag[t], (short)0, acc[m][t],
                    false, false);

#if USE_ASYNC_LDS
        __builtin_amdgcn_s_wait_asynccnt(0);        // my copies landed
#endif
        __syncthreads();                            // everyone's copies visible
#pragma unroll
        for (int m = 0; m < MT; ++m) { alo[m] = alo_n[m]; ahi[m] = ahi_n[m]; }
    }

    // ---- epilogue: C layout lane(0-15)->col, vgpr i -> row (+8 for hi lanes)
#pragma unroll
    for (int m = 0; m < MT; ++m) {
#pragma unroll
        for (int t = 0; t < NT; ++t) {
            const int col = colbase + t * 16 + r;
            const float bv = BIAS ? bias[col] : 0.0f;
#pragma unroll
            for (int i = 0; i < 8; ++i) {
                const int row = rowbase + m * 16 + hiHalf * 8 + i;
                float v = acc[m][t][i] + bv;
                if (RELU) v = (v > 0.0f) ? v : 0.01f * v;  // leaky_relu, slope .01
                if (OUT_F16)
                    ((_Float16*)Cout)[(size_t)row * N + col] = (_Float16)v;
                else
                    ((float*)Cout)[(size_t)row * N + col] = v;
            }
        }
    }
}

// ---------------------------------------------------------------------------
// Fused masked softmax (over 14 atoms) + qh/ph einsums + residue mask.
// One block per residue, 256 threads (one per channel).
// ---------------------------------------------------------------------------
__global__ __launch_bounds__(256)
void attn_pool_kernel(const float*    __restrict__ samout,    // [ATOMS,16]
                      const float*    __restrict__ atom_mask, // [R,14]
                      const _Float16* __restrict__ q16,       // [ATOMS,256]
                      const _Float16* __restrict__ p16,       // [ATOMS,3,256]
                      _Float16* __restrict__ qh,              // [R,2048]
                      _Float16* __restrict__ ph) {            // [R,3,2048]
    __shared__ float s_logit[A14][16];
    __shared__ float s_aw[A14][16];
    __shared__ float s_resmask;

    const int rres = blockIdx.x;
    const int t    = threadIdx.x;

    if (t < A14 * 16) {
        const int a = t >> 4, h = t & 15;
        const float m  = atom_mask[rres * A14 + a];
        const float lm = (1.0f - m + 1e-6f) / (m - 1e-6f);
        s_logit[a][h] = samout[((size_t)rres * A14 + a) * 16 + h] + lm;
    }
    if (t == 0) {
        float any = 0.0f;
        for (int a = 0; a < A14; ++a)
            if (atom_mask[rres * A14 + a] != 0.0f) any = 1.0f;
        s_resmask = any;
    }
    __syncthreads();

    if (t < 16) {
        float mx = -INFINITY;
        for (int a = 0; a < A14; ++a) mx = fmaxf(mx, s_logit[a][t]);
        float e[A14], sum = 0.0f;
        for (int a = 0; a < A14; ++a) { e[a] = expf(s_logit[a][t] - mx); sum += e[a]; }
        const float inv = 1.0f / sum;
        for (int a = 0; a < A14; ++a) s_aw[a][t] = e[a] * inv;
    }
    __syncthreads();

    const float rm = s_resmask;
    const int   c  = t;

    // qh[r, c, h] = sum_a q[r,a,c] * aw[r,a,2h]
    {
        float qv[A14];
#pragma unroll
        for (int a = 0; a < A14; ++a)
            qv[a] = (float)q16[((size_t)rres * A14 + a) * NN0 + c];
#pragma unroll
        for (int h = 0; h < NHH; ++h) {
            float accv = 0.0f;
#pragma unroll
            for (int a = 0; a < A14; ++a) accv += qv[a] * s_aw[a][2 * h];
            qh[(size_t)rres * (NN0 * NHH) + c * NHH + h] = (_Float16)(accv * rm);
        }
    }
    // ph[r, x, c, h] = sum_a p[r,a,x,c] * aw[r,a,2h+1]
    for (int x = 0; x < XX; ++x) {
        float pv[A14];
#pragma unroll
        for (int a = 0; a < A14; ++a)
            pv[a] = (float)p16[(((size_t)rres * A14 + a) * 3 + x) * NN0 + c];
#pragma unroll
        for (int h = 0; h < NHH; ++h) {
            float accv = 0.0f;
#pragma unroll
            for (int a = 0; a < A14; ++a) accv += pv[a] * s_aw[a][2 * h + 1];
            ph[((size_t)rres * 3 + x) * (NN0 * NHH) + c * NHH + h] =
                (_Float16)(accv * rm);
        }
    }
}

// ---------------------------------------------------------------------------
// Host-side orchestration
// ---------------------------------------------------------------------------
extern "C" void kernel_launch(void* const* d_in, const int* in_sizes, int n_in,
                              void* d_out, int out_size, void* d_ws, size_t ws_size,
                              hipStream_t stream) {
    (void)in_sizes; (void)n_in; (void)out_size; (void)ws_size;

    const float* q        = (const float*)d_in[0];
    const float* p        = (const float*)d_in[1];
    const float* mask     = (const float*)d_in[2];
    const float* sam_w1   = (const float*)d_in[3];
    const float* sam_b1   = (const float*)d_in[4];
    const float* sam_w2   = (const float*)d_in[5];
    const float* sam_b2   = (const float*)d_in[6];
    const float* sam_w3   = (const float*)d_in[7];
    const float* sam_b3   = (const float*)d_in[8];
    const float* zdm_w1   = (const float*)d_in[9];
    const float* zdm_b1   = (const float*)d_in[10];
    const float* zdm_w2   = (const float*)d_in[11];
    const float* zdm_b2   = (const float*)d_in[12];
    const float* zdm_w3   = (const float*)d_in[13];
    const float* zdm_b3   = (const float*)d_in[14];
    const float* zdm_vecw = (const float*)d_in[15];

    char* ws = (char*)d_ws;
    size_t off = 0;
    auto take = [&](size_t bytes) { char* ptr = ws + off; off += (bytes + 255) & ~(size_t)255; return ptr; };

    _Float16* z16    = (_Float16*)take((size_t)ATOMS * 512 * 2);
    _Float16* q16    = (_Float16*)take((size_t)ATOMS * NN0 * 2);
    _Float16* p16    = (_Float16*)take((size_t)ATOMS * 3 * NN0 * 2);
    _Float16* h1     = (_Float16*)take((size_t)ATOMS * NN0 * 2);
    _Float16* h2     = (_Float16*)take((size_t)ATOMS * NN0 * 2);
    float*    samout = (float*)   take((size_t)ATOMS * 16 * 4);
    _Float16* qh     = (_Float16*)take((size_t)RR * 2048 * 2);
    _Float16* phbuf  = (_Float16*)take((size_t)RR * 3 * 2048 * 2);
    _Float16* g1     = (_Float16*)take((size_t)RR * NN0 * 2);
    _Float16* g2     = (_Float16*)take((size_t)RR * NN0 * 2);
    _Float16* w1p    = (_Float16*)take((size_t)512 * 256 * 2);
    _Float16* w2p    = (_Float16*)take((size_t)256 * 256 * 2);
    _Float16* w3p    = (_Float16*)take((size_t)256 * 16 * 2);
    _Float16* z1p    = (_Float16*)take((size_t)2048 * 256 * 2);
    _Float16* z2p    = (_Float16*)take((size_t)256 * 256 * 2);
    _Float16* z3p    = (_Float16*)take((size_t)256 * 256 * 2);
    _Float16* vecp   = (_Float16*)take((size_t)2048 * 256 * 2);

    float* qr = (float*)d_out;                       // [8192, 256]
    float* pr = (float*)d_out + (size_t)RR * NN1;    // [8192, 3, 256]

    // ---- weight repacking (tiny) ----
    pack_b_kernel<<<dim3((512 / 32) * (256 / 16)), 32, 0, stream>>>(sam_w1, w1p, 512, 256);
    pack_b_kernel<<<dim3((256 / 32) * (256 / 16)), 32, 0, stream>>>(sam_w2, w2p, 256, 256);
    pack_b_kernel<<<dim3((256 / 32) * (16 / 16)),  32, 0, stream>>>(sam_w3, w3p, 256, 16);
    pack_b_kernel<<<dim3((2048 / 32) * (256 / 16)), 32, 0, stream>>>(zdm_w1, z1p, 2048, 256);
    pack_b_kernel<<<dim3((256 / 32) * (256 / 16)),  32, 0, stream>>>(zdm_w2, z2p, 256, 256);
    pack_b_kernel<<<dim3((256 / 32) * (256 / 16)),  32, 0, stream>>>(zdm_w3, z3p, 256, 256);
    pack_b_kernel<<<dim3((2048 / 32) * (256 / 16)), 32, 0, stream>>>(zdm_vecw, vecp, 2048, 256);

    // ---- prep: z / q / p f16 (single streaming pass over p in f32) ----
    prep_kernel<<<dim3(ATOMS * 64 / 256), 256, 0, stream>>>(q, p, z16, q16, p16);

    // ---- sam MLP over all atoms ----
    wmma_gemm_kernel<2, 4, true, true, true><<<dim3(ATOMS / 256, 4), 256, 0, stream>>>(
        z16, w1p, sam_b1, h1, ATOMS, 512, 256);
    wmma_gemm_kernel<2, 4, true, true, true><<<dim3(ATOMS / 256, 4), 256, 0, stream>>>(
        h1, w2p, sam_b2, h2, ATOMS, 256, 256);
    wmma_gemm_kernel<2, 1, true, false, false><<<dim3(ATOMS / 256, 1), 256, 0, stream>>>(
        h2, w3p, sam_b3, samout, ATOMS, 256, 16);

    // ---- masked softmax + attention pooling (einsums) ----
    attn_pool_kernel<<<dim3(RR), 256, 0, stream>>>(samout, mask, q16, p16, qh, phbuf);

    // ---- zdm MLP (qr) ----
    wmma_gemm_kernel<2, 4, true, true, true><<<dim3(RR / 256, 4), 256, 0, stream>>>(
        qh, z1p, zdm_b1, g1, RR, 2048, 256);
    wmma_gemm_kernel<2, 4, true, true, true><<<dim3(RR / 256, 4), 256, 0, stream>>>(
        g1, z2p, zdm_b2, g2, RR, 256, 256);
    wmma_gemm_kernel<2, 4, true, false, false><<<dim3(RR / 256, 4), 256, 0, stream>>>(
        g2, z3p, zdm_b3, qr, RR, 256, 256);

    // ---- pr = ph_flat @ zdm_vec_w  (M = 8192*3) ----
    wmma_gemm_kernel<2, 4, false, false, false><<<dim3(RR * 3 / 256, 4), 256, 0, stream>>>(
        phbuf, vecp, nullptr, pr, RR * 3, 2048, 256);
}
